// CMuST_DiT_71262097375886
// MI455X (gfx1250) — compile-verified, matrix operand order checked by reading, not run on hardware
//
#include <hip/hip_runtime.h>

// ---------------------------------------------------------------------------
// CDNA5 / gfx1250 DiT block forward.  wave32, WMMA bf16 (f32 accum),
// Tensor Data Mover for A-tile staging into LDS.
// Shapes fixed by the reference: C=384 NH=12 HD=32 MLP=1536 B=8 T=12 N=256
// BT=96, M1 = BT*N = 24576.
// ---------------------------------------------------------------------------

typedef __bf16 bf16;
typedef __attribute__((ext_vector_type(16))) __bf16 v16bf;
typedef __attribute__((ext_vector_type(8)))  __bf16 v8bf;
typedef __attribute__((ext_vector_type(8)))  float  v8f;
typedef __attribute__((ext_vector_type(4)))  float  f32x4;
typedef __attribute__((ext_vector_type(4)))  unsigned int u32x4;
typedef __attribute__((ext_vector_type(8)))  int i32x8;
typedef __attribute__((ext_vector_type(4)))  int i32x4;

#define C_DIM   384
#define NH      12
#define HD      32
#define MLP_DIM 1536
#define BATCH   8
#define TT      12
#define NN      256
#define BT      96
#define M1      24576            // BT*NN
#define Q3      1152             // 3*C
#define MOD9    3456             // 9*C
#define ATT_SCALE 0.17677669529663687f   // 1/sqrt(32)

static __device__ __forceinline__ bf16 f2bf(float f) {
    union { float f; unsigned int u; } in; in.f = f;
    unsigned int u = in.u;
    unsigned int lsb = (u >> 16) & 1u;
    u += 0x7FFFu + lsb;                       // round-to-nearest-even
    union { unsigned short s; bf16 b; } out;
    out.s = (unsigned short)(u >> 16);
    return out.b;
}
static __device__ __forceinline__ float bf2f(bf16 b) {
    union { unsigned short s; bf16 b; } in; in.b = b;
    union { unsigned int u; float f; } out;
    out.u = ((unsigned int)in.s) << 16;
    return out.f;
}

// ---------------------------------------------------------------------------
// TDM: async DMA of a [tileRows x 32] bf16 tile (row stride = strideElems)
// from global into LDS at byte offset lds_off.  D# layout per CDNA5 ISA
// CDNA5 ISA: group0 {count, lds_addr, global_addr, type=2},
// group1 {data_size=2B, tensor_dim0=32, tensor_dim1=rowsValid, tile 32 x
// tileRows, dim0 stride}.  Rows >= rowsValid are hardware zero-filled (OOB
// reads return zero), which handles the M=96 adaLN edge tiles.
// Groups 2/3 zero (2D tensor).  Completion tracked with TENSORcnt.
// This toolchain's builtin takes 6 args (g0, g1, g2, g3, g4, cpol).
// ---------------------------------------------------------------------------
static __device__ __forceinline__ void tdm_load_tile_bf16(
    unsigned int lds_off, const bf16* gptr,
    unsigned int rowsValid, unsigned int tileRows, unsigned int strideElems)
{
    unsigned long long ga = (unsigned long long)(size_t)gptr;
    u32x4 g0;
    g0[0] = 1u;                                            // count=1, user D#
    g0[1] = lds_off;                                       // lds_addr (bytes)
    g0[2] = (unsigned int)(ga & 0xffffffffu);              // global_addr lo
    g0[3] = (unsigned int)((ga >> 32) & 0x01ffffffu)       // global_addr hi
          | (2u << 30);                                    // type = 2 (image)
    i32x8 g1;
    g1[0] = (int)(1u << 16);                               // data_size=1 (2B)
    g1[1] = (int)((32u & 0xffffu) << 16);                  // tensor_dim0 lo16
    g1[2] = (int)(((rowsValid & 0xffffu) << 16));          // td0 hi16=0 | td1 lo16
    g1[3] = (int)(((rowsValid >> 16) & 0xffffu)            // td1 hi16
          | (32u << 16));                                  // tile_dim0 = 32
    g1[4] = (int)(tileRows & 0xffffu);                     // tile_dim1
    g1[5] = (int)strideElems;                              // dim0_stride lo32
    g1[6] = 0;                                             // stride hi / dim1 stride
    g1[7] = 0;
    i32x4 z4 = {};
    i32x8 z8 = {};
    __builtin_amdgcn_tensor_load_to_lds(g0, g1, z4, z4, z8, 0);
}

// ---------------------------------------------------------------------------
// Generic WMMA GEMM:  Y[M,N] = A[M,K](bf16) @ W[N,K](f32->bf16)^T + bias
// Block 256 threads = 8 waves, 64(M) x 128(N) block tile; each wave owns a
// 32x32 piece -> 4 x v_wmma_f32_16x16x32_bf16 per K-step with A/B fragment
// reuse.  A tile staged by the Tensor Data Mover; W tile converted f32->bf16
// through VGPRs.
// epi: 0 = store f32, 1 = store bf16, 2 = store bf16(gelu_exact(v))
// ---------------------------------------------------------------------------
__global__ __launch_bounds__(256)
void gemm_bf16_wmma(const bf16* __restrict__ A, const float* __restrict__ W,
                    const float* __restrict__ bias,
                    float* __restrict__ Yf, bf16* __restrict__ Yb,
                    int M, int N, int K, int epi)
{
    __shared__ __align__(16) bf16 As[64][40];
    __shared__ __align__(16) bf16 Bs[128][40];

    const int tid    = threadIdx.x;
    const int lane   = tid & 31;
    const int wave   = tid >> 5;
    const int mBase  = blockIdx.y * 64;
    const int nBase  = blockIdx.x * 128;
    const int rowsub = (wave >> 2) * 32;      // 0 or 32
    const int colsub = (wave & 3) * 32;       // 0,32,64,96

    const int wr_ = tid >> 1;                 // 0..127 (W tile row)
    const int wc_ = (tid & 1) * 16;           // 0 or 16 (W tile col chunk)

    const int fr = lane & 15;
    const int kb = (lane >> 4) * 8;

    const unsigned int rowsValid = (unsigned int)(M - mBase);

    v8f acc00 = {}; v8f acc01 = {}; v8f acc10 = {}; v8f acc11 = {};

    for (int k0 = 0; k0 < K; k0 += 32) {
        // ---- A tile via Tensor Data Mover (one issue per block) ----
        if (wave == 0) {
            tdm_load_tile_bf16((unsigned int)(size_t)&As[0][0],
                               A + (size_t)mBase * K + k0,
                               rowsValid, 64u, (unsigned int)K);
        }
        // ---- W tile: global f32 -> bf16 -> LDS (16 elems / thread) ----
        {
            int wr = nBase + wr_; if (wr >= N) wr = N - 1;
            const float* wp = W + (size_t)wr * K + k0 + wc_;
            f32x4 w0 = *(const f32x4*)(wp);
            f32x4 w1 = *(const f32x4*)(wp + 4);
            f32x4 w2 = *(const f32x4*)(wp + 8);
            f32x4 w3 = *(const f32x4*)(wp + 12);
            v8bf p0, p1;
            p0[0] = f2bf(w0[0]); p0[1] = f2bf(w0[1]);
            p0[2] = f2bf(w0[2]); p0[3] = f2bf(w0[3]);
            p0[4] = f2bf(w1[0]); p0[5] = f2bf(w1[1]);
            p0[6] = f2bf(w1[2]); p0[7] = f2bf(w1[3]);
            p1[0] = f2bf(w2[0]); p1[1] = f2bf(w2[1]);
            p1[2] = f2bf(w2[2]); p1[3] = f2bf(w2[3]);
            p1[4] = f2bf(w3[0]); p1[5] = f2bf(w3[1]);
            p1[6] = f2bf(w3[2]); p1[7] = f2bf(w3[3]);
            *(v8bf*)(&Bs[wr_][wc_])     = p0;
            *(v8bf*)(&Bs[wr_][wc_ + 8]) = p1;
            if (k0 + 32 < K)
                __builtin_prefetch(wp + 32, 0, 0);
        }
        if (wave == 0) __builtin_amdgcn_s_wait_tensorcnt(0);
        __syncthreads();

        // ---- fragments (ISA 7.12.2: lanes<16 K=kb..kb+7,kb+16..kb+23) ----
        v16bf a0, a1, b0, b1;
        #pragma unroll
        for (int i = 0; i < 8; i++) {
            a0[i]     = As[rowsub + fr][kb + i];
            a0[8 + i] = As[rowsub + fr][kb + 16 + i];
            a1[i]     = As[rowsub + 16 + fr][kb + i];
            a1[8 + i] = As[rowsub + 16 + fr][kb + 16 + i];
            b0[i]     = Bs[colsub + fr][kb + i];
            b0[8 + i] = Bs[colsub + fr][kb + 16 + i];
            b1[i]     = Bs[colsub + 16 + fr][kb + i];
            b1[8 + i] = Bs[colsub + 16 + fr][kb + 16 + i];
        }
        acc00 = __builtin_amdgcn_wmma_f32_16x16x32_bf16(false, a0, false, b0,
                                                        (short)0, acc00, false, false);
        acc01 = __builtin_amdgcn_wmma_f32_16x16x32_bf16(false, a0, false, b1,
                                                        (short)0, acc01, false, false);
        acc10 = __builtin_amdgcn_wmma_f32_16x16x32_bf16(false, a1, false, b0,
                                                        (short)0, acc10, false, false);
        acc11 = __builtin_amdgcn_wmma_f32_16x16x32_bf16(false, a1, false, b1,
                                                        (short)0, acc11, false, false);
        __syncthreads();
    }

    // ---- epilogue ----
    const int c0   = nBase + colsub + fr;
    const int c1   = c0 + 16;
    const float bv0 = bias[c0];
    const float bv1 = bias[c1];
    const int rtop = mBase + rowsub + ((lane >> 4) * 8);
    #pragma unroll
    for (int i = 0; i < 8; i++) {
        int r0 = rtop + i;
        int r1 = rtop + 16 + i;
        float v00 = acc00[i] + bv0, v01 = acc01[i] + bv1;
        float v10 = acc10[i] + bv0, v11 = acc11[i] + bv1;
        if (epi == 2) {
            v00 = 0.5f * v00 * (1.0f + erff(v00 * 0.7071067811865476f));
            v01 = 0.5f * v01 * (1.0f + erff(v01 * 0.7071067811865476f));
            v10 = 0.5f * v10 * (1.0f + erff(v10 * 0.7071067811865476f));
            v11 = 0.5f * v11 * (1.0f + erff(v11 * 0.7071067811865476f));
        }
        if (r0 < M) {
            if (epi == 0) { Yf[(size_t)r0 * N + c0] = v00; Yf[(size_t)r0 * N + c1] = v01; }
            else          { Yb[(size_t)r0 * N + c0] = f2bf(v00); Yb[(size_t)r0 * N + c1] = f2bf(v01); }
        }
        if (r1 < M) {
            if (epi == 0) { Yf[(size_t)r1 * N + c0] = v10; Yf[(size_t)r1 * N + c1] = v11; }
            else          { Yb[(size_t)r1 * N + c0] = f2bf(v10); Yb[(size_t)r1 * N + c1] = f2bf(v11); }
        }
    }
}

// ---------------------------------------------------------------------------
// silu(c) -> bf16
// ---------------------------------------------------------------------------
__global__ __launch_bounds__(256)
void silu_kernel(const float* __restrict__ c, bf16* __restrict__ out, int n)
{
    int i = blockIdx.x * 256 + threadIdx.x;
    if (i < n) {
        float v = c[i];
        out[i] = f2bf(v / (1.0f + __expf(-v)));
    }
}

// ---------------------------------------------------------------------------
// LayerNorm(eps=1e-6, no affine) + adaLN modulate -> bf16.
// One wave per row (384 cols, 12 per lane).
// ---------------------------------------------------------------------------
__global__ __launch_bounds__(256)
void lnmod_kernel(const float* __restrict__ x, const float* __restrict__ mod,
                  bf16* __restrict__ out, int scaleOff, int shiftOff)
{
    const int row  = blockIdx.x * 8 + (threadIdx.x >> 5);
    const int lane = threadIdx.x & 31;
    const float* xr = x + (size_t)row * C_DIM;

    float v[12]; float s = 0.f, s2 = 0.f;
    #pragma unroll
    for (int k = 0; k < 12; k++) {
        float t = xr[lane + 32 * k];
        v[k] = t; s += t; s2 += t * t;
    }
    #pragma unroll
    for (int o = 16; o > 0; o >>= 1) {
        s  += __shfl_xor(s,  o);
        s2 += __shfl_xor(s2, o);
    }
    float mu   = s  * (1.0f / 384.0f);
    float var  = s2 * (1.0f / 384.0f) - mu * mu;
    float rstd = rsqrtf(var + 1e-6f);

    const float* mrow = mod + (size_t)(row >> 8) * MOD9;   // bt = row / 256
    bf16* orow = out + (size_t)row * C_DIM;
    #pragma unroll
    for (int k = 0; k < 12; k++) {
        int cidx = lane + 32 * k;
        float sc = mrow[scaleOff + cidx];
        float sh = mrow[shiftOff + cidx];
        orow[cidx] = f2bf((v[k] - mu) * rstd * (1.0f + sc) + sh);
    }
}

// ---------------------------------------------------------------------------
// S = scale * Q@K^T + rel_pos_bias  -> attn_map (f32, pre-softmax)
// One 16x16 tile per wave via one v_wmma_f32_16x16x32_bf16 (K=HD=32).
// grid = (BT*NH, 16 row-tiles, 2), block 256 (waves cover 8 col-tiles each z).
// ---------------------------------------------------------------------------
__global__ __launch_bounds__(256)
void qk_bias_kernel(const bf16* __restrict__ qkv, const float* __restrict__ bias,
                    float* __restrict__ amap)
{
    const int bh   = blockIdx.x;          // bt*12 + h
    const int bt   = bh / NH, h = bh % NH;
    const int rt   = blockIdx.y;
    const int lane = threadIdx.x & 31;
    const int wave = threadIdx.x >> 5;
    const int ct   = blockIdx.z * 8 + wave;
    const int fr   = lane & 15;
    const int kb   = (lane >> 4) * 8;

    const bf16* qp = qkv + (size_t)(bt * NN + rt * 16 + fr) * Q3 + h * HD;
    const bf16* kp = qkv + (size_t)(bt * NN + ct * 16 + fr) * Q3 + C_DIM + h * HD;
    v8bf qlo = *(const v8bf*)(qp + kb);
    v8bf qhi = *(const v8bf*)(qp + kb + 16);
    v8bf klo = *(const v8bf*)(kp + kb);
    v8bf khi = *(const v8bf*)(kp + kb + 16);

    v16bf a, b;
    #pragma unroll
    for (int i = 0; i < 8; i++) {
        a[i] = qlo[i]; a[8 + i] = qhi[i];
        b[i] = klo[i]; b[8 + i] = khi[i];
    }
    v8f acc = {};
    acc = __builtin_amdgcn_wmma_f32_16x16x32_bf16(false, a, false, b,
                                                  (short)0, acc, false, false);

    const int col   = ct * 16 + fr;
    const int rtop  = rt * 16 + ((lane >> 4) * 8);
    const float* bp = bias + (size_t)h * NN * NN;
    float* op       = amap + (size_t)bh * NN * NN;
    #pragma unroll
    for (int i = 0; i < 8; i++) {
        int r = rtop + i;
        op[(size_t)r * NN + col] = acc[i] * ATT_SCALE + bp[(size_t)r * NN + col];
    }
}

// ---------------------------------------------------------------------------
// In-place row softmax over attn_map (rows of 256).  One wave per row.
// ---------------------------------------------------------------------------
__global__ __launch_bounds__(256)
void softmax256_kernel(float* __restrict__ amap)
{
    const int row  = blockIdx.x * 8 + (threadIdx.x >> 5);
    const int lane = threadIdx.x & 31;
    float* p = amap + (size_t)row * NN;

    float v[8]; float mx = -1e30f;
    #pragma unroll
    for (int k = 0; k < 8; k++) { v[k] = p[lane + 32 * k]; mx = fmaxf(mx, v[k]); }
    #pragma unroll
    for (int o = 16; o > 0; o >>= 1) mx = fmaxf(mx, __shfl_xor(mx, o));
    float s = 0.f;
    #pragma unroll
    for (int k = 0; k < 8; k++) { v[k] = __expf(v[k] - mx); s += v[k]; }
    #pragma unroll
    for (int o = 16; o > 0; o >>= 1) s += __shfl_xor(s, o);
    float inv = 1.0f / s;
    #pragma unroll
    for (int k = 0; k < 8; k++) p[lane + 32 * k] = v[k] * inv;
}

// ---------------------------------------------------------------------------
// O = P @ V   (P read back f32 from attn_map, V from qkv buffer).
// M=256 rows x N=32 cols per (bt,h); K=256 -> 8 WMMA per tile.
// grid = (BT*NH, 4), block 256: 8 tiles per block, 32 tiles per (bt,h).
// Output written bf16 into the [M1, C] layout at col h*32+d.
// ---------------------------------------------------------------------------
__global__ __launch_bounds__(256)
void pv_kernel(const float* __restrict__ amap, const bf16* __restrict__ qkv,
               bf16* __restrict__ o)
{
    const int bh   = blockIdx.x;
    const int bt   = bh / NH, h = bh % NH;
    const int lane = threadIdx.x & 31;
    const int wave = threadIdx.x >> 5;
    const int tile = blockIdx.y * 8 + wave;   // 0..31
    const int rt   = tile >> 1;               // 0..15
    const int ct   = tile & 1;                // 0..1
    const int fr   = lane & 15;
    const int kb   = (lane >> 4) * 8;

    const float* prow = amap + ((size_t)bh * NN + rt * 16 + fr) * NN;
    const bf16* vcol  = qkv + 2 * C_DIM + h * HD + ct * 16 + fr;   // + j*Q3

    v8f acc = {};
    for (int kk = 0; kk < NN; kk += 32) {
        v16bf a, b;
        #pragma unroll
        for (int i = 0; i < 8; i++) {
            a[i]     = f2bf(prow[kk + kb + i]);
            a[8 + i] = f2bf(prow[kk + kb + 16 + i]);
            b[i]     = vcol[(size_t)(bt * NN + kk + kb + i) * Q3];
            b[8 + i] = vcol[(size_t)(bt * NN + kk + kb + 16 + i) * Q3];
        }
        acc = __builtin_amdgcn_wmma_f32_16x16x32_bf16(false, a, false, b,
                                                      (short)0, acc, false, false);
    }
    const int col  = h * HD + ct * 16 + fr;
    const int rtop = rt * 16 + ((lane >> 4) * 8);
    bf16* op = o + (size_t)(bt * NN) * C_DIM;
    #pragma unroll
    for (int i = 0; i < 8; i++)
        op[(size_t)(rtop + i) * C_DIM + col] = f2bf(acc[i]);
}

// ---------------------------------------------------------------------------
// Temporal attention core (T=12, tiny): one wave per (b, n, h).
// lane = head dim d.  Cross-lane dots via wave32 shfl_xor reduction.
// ---------------------------------------------------------------------------
__global__ __launch_bounds__(32)
void tattn_kernel(const bf16* __restrict__ qkv, bf16* __restrict__ o)
{
    const int bn   = blockIdx.x;        // b*256 + n
    const int h    = blockIdx.y;
    const int lane = threadIdx.x;       // d
    const int b    = bn >> 8, n = bn & 255;

    float q[TT], k[TT], vv[TT];
    #pragma unroll
    for (int t = 0; t < TT; t++) {
        size_t m = (size_t)((b * TT + t) * NN + n) * Q3 + h * HD + lane;
        q[t]  = bf2f(qkv[m]);
        k[t]  = bf2f(qkv[m + C_DIM]);
        vv[t] = bf2f(qkv[m + 2 * C_DIM]);
    }
    #pragma unroll
    for (int i = 0; i < TT; i++) {
        float s[TT]; float mx = -1e30f;
        #pragma unroll
        for (int j = 0; j < TT; j++) {
            float p = q[i] * k[j];
            #pragma unroll
            for (int off = 16; off > 0; off >>= 1) p += __shfl_xor(p, off);
            s[j] = p * ATT_SCALE;
            mx = fmaxf(mx, s[j]);
        }
        float sum = 0.f;
        #pragma unroll
        for (int j = 0; j < TT; j++) { s[j] = __expf(s[j] - mx); sum += s[j]; }
        float inv = 1.0f / sum;
        float outv = 0.f;
        #pragma unroll
        for (int j = 0; j < TT; j++) outv += s[j] * vv[j];
        o[(size_t)((b * TT + i) * NN + n) * C_DIM + h * HD + lane] = f2bf(outv * inv);
    }
}

// ---------------------------------------------------------------------------
// x += gate[bt] * y   (y already includes the proj/mlp bias)
// ---------------------------------------------------------------------------
__global__ __launch_bounds__(256)
void residual_kernel(float* __restrict__ x, const float* __restrict__ y,
                     const float* __restrict__ mod, int gateOff)
{
    size_t idx = (size_t)blockIdx.x * 256 + threadIdx.x;   // < M1*C
    int cidx = (int)(idx % C_DIM);
    int bt   = (int)((idx / C_DIM) >> 8);
    float g  = mod[(size_t)bt * MOD9 + gateOff + cidx];
    x[idx] += g * y[idx];
}

// ---------------------------------------------------------------------------
// Host side
// ---------------------------------------------------------------------------
extern "C" void kernel_launch(void* const* d_in, const int* in_sizes, int n_in,
                              void* d_out, int out_size, void* d_ws, size_t ws_size,
                              hipStream_t stream)
{
    (void)in_sizes; (void)n_in; (void)out_size; (void)ws_size;

    const float* x        = (const float*)d_in[0];
    const float* c        = (const float*)d_in[1];
    const float* rpb      = (const float*)d_in[2];
    const float* w_adaln  = (const float*)d_in[3];
    const float* b_adaln  = (const float*)d_in[4];
    const float* wqkv_s   = (const float*)d_in[5];
    const float* bqkv_s   = (const float*)d_in[6];
    const float* wproj_s  = (const float*)d_in[7];
    const float* bproj_s  = (const float*)d_in[8];
    const float* wqkv_t   = (const float*)d_in[9];
    const float* bqkv_t   = (const float*)d_in[10];
    const float* wproj_t  = (const float*)d_in[11];
    const float* bproj_t  = (const float*)d_in[12];
    const float* w1       = (const float*)d_in[13];
    const float* b1       = (const float*)d_in[14];
    const float* w2       = (const float*)d_in[15];
    const float* b2       = (const float*)d_in[16];
    // d_in[17] = T_pred (== 12, compile-time here)

    float* xres = (float*)d_out;                       // [M1, C]
    float* amap = xres + (size_t)M1 * C_DIM;           // [BT, NH, 256, 256]

    // workspace carve (~134 MB total)
    char* wp = (char*)d_ws;
    size_t off = 0;
    auto carve = [&](size_t bytes) -> void* {
        void* p = wp + off; off += (bytes + 255) & ~(size_t)255; return p;
    };
    float* mod   = (float*)carve((size_t)BT * MOD9 * 4);           // 1.3 MB
    bf16*  siluc = (bf16*) carve((size_t)BT * C_DIM * 2);          // 74 KB
    bf16*  bufA  = (bf16*) carve((size_t)M1 * MLP_DIM * 2);        // 75.5 MB (qkv / mlp-h)
    bf16*  bufB  = (bf16*) carve((size_t)M1 * C_DIM * 2);          // 18.9 MB (xmod / o)
    float* bufC  = (float*)carve((size_t)M1 * C_DIM * 4);          // 37.7 MB (gemm f32 out)

    const dim3 blk256(256);

    // 1) adaLN: mod = silu(c) @ w_adaln^T + b_adaln
    silu_kernel<<<dim3((BT * C_DIM + 255) / 256), blk256, 0, stream>>>(c, siluc, BT * C_DIM);
    gemm_bf16_wmma<<<dim3(MOD9 / 128, (BT + 63) / 64), blk256, 0, stream>>>(
        siluc, w_adaln, b_adaln, mod, nullptr, BT, MOD9, C_DIM, 0);

    // 2) residual accumulator = x
    (void)hipMemcpyAsync(xres, x, (size_t)M1 * C_DIM * 4, hipMemcpyDeviceToDevice, stream);

    // ---------------- spatial attention ----------------
    lnmod_kernel<<<dim3(M1 / 8), blk256, 0, stream>>>(xres, mod, bufB, 1 * C_DIM, 0 * C_DIM);
    gemm_bf16_wmma<<<dim3(Q3 / 128, M1 / 64), blk256, 0, stream>>>(
        bufB, wqkv_s, bqkv_s, nullptr, bufA, M1, Q3, C_DIM, 1);
    qk_bias_kernel<<<dim3(BT * NH, 16, 2), blk256, 0, stream>>>(bufA, rpb, amap);
    softmax256_kernel<<<dim3(BT * NH * NN / 8), blk256, 0, stream>>>(amap);
    pv_kernel<<<dim3(BT * NH, 4), blk256, 0, stream>>>(amap, bufA, bufB);
    gemm_bf16_wmma<<<dim3(C_DIM / 128, M1 / 64), blk256, 0, stream>>>(
        bufB, wproj_s, bproj_s, bufC, nullptr, M1, C_DIM, C_DIM, 0);
    residual_kernel<<<dim3((unsigned)((size_t)M1 * C_DIM / 256)), blk256, 0, stream>>>(
        xres, bufC, mod, 2 * C_DIM);

    // ---------------- temporal attention ----------------
    lnmod_kernel<<<dim3(M1 / 8), blk256, 0, stream>>>(xres, mod, bufB, 4 * C_DIM, 3 * C_DIM);
    gemm_bf16_wmma<<<dim3(Q3 / 128, M1 / 64), blk256, 0, stream>>>(
        bufB, wqkv_t, bqkv_t, nullptr, bufA, M1, Q3, C_DIM, 1);
    tattn_kernel<<<dim3(BATCH * NN, NH), dim3(32), 0, stream>>>(bufA, bufB);
    gemm_bf16_wmma<<<dim3(C_DIM / 128, M1 / 64), blk256, 0, stream>>>(
        bufB, wproj_t, bproj_t, bufC, nullptr, M1, C_DIM, C_DIM, 0);
    residual_kernel<<<dim3((unsigned)((size_t)M1 * C_DIM / 256)), blk256, 0, stream>>>(
        xres, bufC, mod, 5 * C_DIM);

    // ---------------- MLP ----------------
    lnmod_kernel<<<dim3(M1 / 8), blk256, 0, stream>>>(xres, mod, bufB, 7 * C_DIM, 6 * C_DIM);
    gemm_bf16_wmma<<<dim3(MLP_DIM / 128, M1 / 64), blk256, 0, stream>>>(
        bufB, w1, b1, nullptr, bufA, M1, MLP_DIM, C_DIM, 2);           // fused exact GELU
    gemm_bf16_wmma<<<dim3(C_DIM / 128, M1 / 64), blk256, 0, stream>>>(
        bufA, w2, b2, bufC, nullptr, M1, C_DIM, MLP_DIM, 0);
    residual_kernel<<<dim3((unsigned)((size_t)M1 * C_DIM / 256)), blk256, 0, stream>>>(
        xres, bufC, mod, 8 * C_DIM);
}
